// GraphAttentionLayer_47708496724382
// MI455X (gfx1250) — matrix operand order repaired and checked by reading, not compiled, not run on hardware
//
#include <hip/hip_runtime.h>
#include <hip/hip_bf16.h>
#include <math.h>

typedef _Float16 v16h __attribute__((ext_vector_type(16)));
typedef float    v8f  __attribute__((ext_vector_type(8)));
typedef int      rvi4 __attribute__((vector_size(16)));   // raw vector for async builtin

#define N_NODES 4096
#define F_IN    256
#define F_OUT   64
#define H_HEADS 4
#define NEG_BIG_F (-9000000000000000.0f)

#if __has_builtin(__builtin_amdgcn_global_load_async_to_lds_b128) && \
    __has_builtin(__builtin_amdgcn_s_wait_asynccnt)
#define USE_ASYNC_LDS 1
#else
#define USE_ASYNC_LDS 0
#endif

// -----------------------------------------------------------------------------
// Kernel 1: Wh = x @ W (fp32 — accuracy-critical, feeds exp via s1/s2).
// Emits s1[h][n], s2[h][n] (fp32) and WhT_f16[h][o][n] (WMMA B operand).
// -----------------------------------------------------------------------------
__global__ __launch_bounds__(64)
void gat_wh_kernel(const float* __restrict__ x, const float* __restrict__ W,
                   const float* __restrict__ a,
                   _Float16* __restrict__ whT,
                   float* __restrict__ s1, float* __restrict__ s2) {
    const int n = blockIdx.x;
    const int h = blockIdx.y;
    const int o = threadIdx.x;           // 0..63

    __shared__ float xs[F_IN];
    __shared__ float red1[64];
    __shared__ float red2[64];

    ((float4*)xs)[o] = ((const float4*)(x + (size_t)n * F_IN))[o];
    __syncthreads();

    float acc = 0.0f;
    const float* wp = W + ((size_t)h * F_IN) * F_OUT + o;
#pragma unroll 8
    for (int f = 0; f < F_IN; ++f) {
        acc = fmaf(xs[f], wp[(size_t)f * F_OUT], acc);
    }

    whT[((size_t)(h * F_OUT + o)) * N_NODES + n] = (_Float16)acc;

    red1[o] = acc * a[h * (2 * F_OUT) + o];
    red2[o] = acc * a[h * (2 * F_OUT) + F_OUT + o];
    __syncthreads();
#pragma unroll
    for (int s = 32; s > 0; s >>= 1) {
        if (o < s) { red1[o] += red1[o + s]; red2[o] += red2[o + s]; }
        __syncthreads();
    }
    if (o == 0) {
        s1[h * N_NODES + n] = red1[0];
        s2[h * N_NODES + n] = red2[0];
    }
}

// -----------------------------------------------------------------------------
// Kernel 2a: exact per-row softmax max for all 4 heads in one adj pass.
// leaky_relu monotonic => max_{adj=1} leaky(s1_i+s2_j) = leaky(s1_i + max s2_j).
// -----------------------------------------------------------------------------
__global__ __launch_bounds__(256)
void gat_rowmax_kernel(const int* __restrict__ adj,
                       const float* __restrict__ s1g,
                       const float* __restrict__ s2g,
                       float* __restrict__ Mrow) {
    const int i = blockIdx.x;
    const int t = threadIdx.x;
    const int j0 = t * 16;

    __shared__ float red[H_HEADS * 256];

    const int* ar = adj + (size_t)i * N_NODES + j0;
    int4 a0 = ((const int4*)ar)[0];
    int4 a1 = ((const int4*)ar)[1];
    int4 a2 = ((const int4*)ar)[2];
    int4 a3 = ((const int4*)ar)[3];
    int mk[16] = { a0.x, a0.y, a0.z, a0.w, a1.x, a1.y, a1.z, a1.w,
                   a2.x, a2.y, a2.z, a2.w, a3.x, a3.y, a3.z, a3.w };

#pragma unroll
    for (int h = 0; h < H_HEADS; ++h) {
        const float* sp = s2g + h * N_NODES + j0;
        float4 f0 = ((const float4*)sp)[0];
        float4 f1 = ((const float4*)sp)[1];
        float4 f2 = ((const float4*)sp)[2];
        float4 f3 = ((const float4*)sp)[3];
        float sv[16] = { f0.x, f0.y, f0.z, f0.w, f1.x, f1.y, f1.z, f1.w,
                         f2.x, f2.y, f2.z, f2.w, f3.x, f3.y, f3.z, f3.w };
        float pm = -INFINITY;
#pragma unroll
        for (int k = 0; k < 16; ++k) {
            pm = fmaxf(pm, (mk[k] > 0) ? sv[k] : -INFINITY);
        }
        red[h * 256 + t] = pm;
    }
    __syncthreads();
#pragma unroll
    for (int s = 128; s > 0; s >>= 1) {
        if (t < s) {
            red[0 * 256 + t] = fmaxf(red[0 * 256 + t], red[0 * 256 + t + s]);
            red[1 * 256 + t] = fmaxf(red[1 * 256 + t], red[1 * 256 + t + s]);
            red[2 * 256 + t] = fmaxf(red[2 * 256 + t], red[2 * 256 + t + s]);
            red[3 * 256 + t] = fmaxf(red[3 * 256 + t], red[3 * 256 + t + s]);
        }
        __syncthreads();
    }
    if (t < H_HEADS) {
        float s2m = red[t * 256];
        float mv;
        if (s2m == -INFINITY) {
            mv = NEG_BIG_F;
        } else {
            float u = s1g[t * N_NODES + i] + s2m;
            mv = (u > 0.0f) ? u : 0.2f * u;
        }
        Mrow[t * N_NODES + i] = mv;
    }
}

// -----------------------------------------------------------------------------
// Kernel 2b: fused masked softmax + PV GEMM via WMMA.
// Block = 128 threads = 4 waves; wave w = head w; 16 query rows per block.
// adj tiles double-buffered in LDS, staged by GLOBAL_LOAD_ASYNC_TO_LDS_B128
// (ASYNCcnt) when available; one barrier per iteration.
// -----------------------------------------------------------------------------
__global__ __launch_bounds__(128)
void gat_attn_kernel(const int* __restrict__ adj,
                     const _Float16* __restrict__ whT,
                     const float* __restrict__ s1g,
                     const float* __restrict__ s2g,
                     const float* __restrict__ Mrow,
                     float* __restrict__ out) {
    __shared__ int lds_adj[2][16 * 36];   // 16x32 tiles, rows padded to 36

    const int i0    = blockIdx.x * 16;
    const int tid   = threadIdx.x;
    const int h     = tid >> 5;           // head = wave id
    const int lane  = tid & 31;
    const int r     = lane & 15;
    const int khalf = lane >> 4;

    const int ldrow  = tid >> 3;          // 0..15
    const int ldquad = (tid & 7) << 2;    // 0..28
    const int* gsrc  = adj + (size_t)(i0 + ldrow) * N_NODES + ldquad;
    const int ldsoff = ldrow * 36 + ldquad;

    const float s1v = s1g[h * N_NODES + i0 + r];
    const float Mv  = Mrow[h * N_NODES + i0 + r];

    float l0 = 0.0f, l1 = 0.0f, l2 = 0.0f, l3 = 0.0f;  // split row-sum chains
    v8f acc0 = {}, acc1 = {}, acc2 = {}, acc3 = {};

    // ---- preload tile 0 into buffer 0 ----
#if USE_ASYNC_LDS
    __builtin_amdgcn_global_load_async_to_lds_b128(
        (rvi4*)gsrc, (rvi4*)&lds_adj[0][ldsoff], 0, 0);
#else
    {
        int4 at = *(const int4*)(gsrc);
        *(int4*)(&lds_adj[0][ldsoff]) = at;
    }
#endif

    const int NT = N_NODES / 32;
    for (int jt = 0; jt < NT; ++jt) {
        const int j0  = jt * 32;
        const int buf = jt & 1;

#if USE_ASYNC_LDS
        __builtin_amdgcn_s_wait_asynccnt(0);   // own LDS writes landed
        __syncthreads();                       // everyone's landed; buf^1 free
        if (jt + 1 < NT) {
            __builtin_amdgcn_global_load_async_to_lds_b128(
                (rvi4*)(gsrc + (jt + 1) * 32),
                (rvi4*)&lds_adj[buf ^ 1][ldsoff], 0, 0);
            __builtin_prefetch(gsrc + (jt + 2) * 32, 0, 0);
        }
#else
        __syncthreads();
        int4 nx;
        if (jt + 1 < NT) {
            nx = *(const int4*)(gsrc + (jt + 1) * 32);
            __builtin_prefetch(gsrc + (jt + 2) * 32, 0, 0);
        }
#endif

        // ---- s2 for this lane's 16 columns ----
        const float* s2p = s2g + h * N_NODES + j0 + 8 * khalf;
        float4 sA = *(const float4*)(s2p);
        float4 sB = *(const float4*)(s2p + 4);
        float4 sC = *(const float4*)(s2p + 16);
        float4 sD = *(const float4*)(s2p + 20);
        float s2v[16] = { sA.x, sA.y, sA.z, sA.w, sB.x, sB.y, sB.z, sB.w,
                          sC.x, sC.y, sC.z, sC.w, sD.x, sD.y, sD.z, sD.w };

        // ---- adj mask from LDS ----
        const int ab = r * 36 + 8 * khalf;
        int4 mA = *(const int4*)(&lds_adj[buf][ab]);
        int4 mB = *(const int4*)(&lds_adj[buf][ab + 4]);
        int4 mC = *(const int4*)(&lds_adj[buf][ab + 16]);
        int4 mD = *(const int4*)(&lds_adj[buf][ab + 20]);
        int mk[16] = { mA.x, mA.y, mA.z, mA.w, mB.x, mB.y, mB.z, mB.w,
                       mC.x, mC.y, mC.z, mC.w, mD.x, mD.y, mD.z, mD.w };

        // ---- p = exp(masked leaky(s1+s2) - Mrow); 4 independent sum chains ----
        float ps[16];
        v16h pa;   // 16-bit A-matrix 16x32 wave32 layout
#pragma unroll
        for (int k = 0; k < 16; ++k) {
            float t = s1v + s2v[k];
            t = (t > 0.0f) ? t : 0.2f * t;
            t = (mk[k] > 0) ? t : NEG_BIG_F;
            float p = __expf(t - Mv);
            ps[k] = p;
            pa[k] = (_Float16)p;
        }
#pragma unroll
        for (int k = 0; k < 4; ++k) {
            l0 += ps[k];
            l1 += ps[4 + k];
            l2 += ps[8 + k];
            l3 += ps[12 + k];
        }

        // ---- B tiles (WhT f16: lane r = N col, K = 8*khalf + s (s<8), 16 + 8*khalf + (s-8) (s>=8)) ----
        const _Float16* wb = whT + ((size_t)(h * F_OUT + r)) * N_NODES + j0 + 8 * khalf;
        union BU { float4 f[2]; v16h hv; } b0, b1, b2, b3;
        b0.f[0] = *(const float4*)(wb);
        b0.f[1] = *(const float4*)(wb + 16);
        b1.f[0] = *(const float4*)(wb + 16 * (size_t)N_NODES);
        b1.f[1] = *(const float4*)(wb + 16 * (size_t)N_NODES + 16);
        b2.f[0] = *(const float4*)(wb + 32 * (size_t)N_NODES);
        b2.f[1] = *(const float4*)(wb + 32 * (size_t)N_NODES + 16);
        b3.f[0] = *(const float4*)(wb + 48 * (size_t)N_NODES);
        b3.f[1] = *(const float4*)(wb + 48 * (size_t)N_NODES + 16);

        acc0 = __builtin_amdgcn_wmma_f32_16x16x32_f16(false, pa, false, b0.hv,
                                                      (short)0, acc0, false, false);
        acc1 = __builtin_amdgcn_wmma_f32_16x16x32_f16(false, pa, false, b1.hv,
                                                      (short)0, acc1, false, false);
        acc2 = __builtin_amdgcn_wmma_f32_16x16x32_f16(false, pa, false, b2.hv,
                                                      (short)0, acc2, false, false);
        acc3 = __builtin_amdgcn_wmma_f32_16x16x32_f16(false, pa, false, b3.hv,
                                                      (short)0, acc3, false, false);

#if !USE_ASYNC_LDS
        if (jt + 1 < NT) {
            *(int4*)(&lds_adj[buf ^ 1][ldsoff]) = nx;
        }
#endif
    }

    // ---- epilogue: combine sum chains, cross-lane reduce, normalize, store ----
    float lsum = (l0 + l1) + (l2 + l3);
    lsum += __shfl_xor(lsum, 16, 32);
#pragma unroll
    for (int g = 0; g < 8; ++g) {
        float lg  = __shfl(lsum, g + 8 * khalf, 32);
        float inv = __builtin_amdgcn_rcpf(lg);
        const int row = i0 + g + 8 * khalf;
        float* op = out + (size_t)row * (H_HEADS * F_OUT) + h * F_OUT + r;
        op[0]  = acc0[g] * inv;
        op[16] = acc1[g] * inv;
        op[32] = acc2[g] * inv;
        op[48] = acc3[g] * inv;
    }
}

// -----------------------------------------------------------------------------
// Host launcher
// -----------------------------------------------------------------------------
extern "C" void kernel_launch(void* const* d_in, const int* in_sizes, int n_in,
                              void* d_out, int out_size, void* d_ws, size_t ws_size,
                              hipStream_t stream) {
    const float* x   = (const float*)d_in[0];
    const int*   adj = (const int*)  d_in[1];
    const float* W   = (const float*)d_in[2];
    const float* a   = (const float*)d_in[3];
    float*       out = (float*)d_out;

    // workspace: WhT f16 (2 MB) | s1 | s2 | Mrow  (fp32, 64 KB each)
    _Float16* whT = (_Float16*)d_ws;
    float* s1   = (float*)((char*)d_ws + (size_t)H_HEADS * F_OUT * N_NODES * sizeof(_Float16));
    float* s2   = s1 + (size_t)H_HEADS * N_NODES;
    float* Mrow = s2 + (size_t)H_HEADS * N_NODES;

    gat_wh_kernel<<<dim3(N_NODES, H_HEADS), 64, 0, stream>>>(x, W, a, whT, s1, s2);
    gat_rowmax_kernel<<<dim3(N_NODES), 256, 0, stream>>>(adj, s1, s2, Mrow);
    gat_attn_kernel<<<dim3(N_NODES / 16), 128, 0, stream>>>(adj, whT, s1, s2, Mrow, out);
}
